// RelationshipModule_28295244546254
// MI455X (gfx1250) — compile-verified, matrix operand order checked by reading, not run on hardware
//
#include <hip/hip_runtime.h>
#include <hip/hip_bf16.h>
#include <cstdint>

// ---------------------------------------------------------------------------
// MI455X (gfx1250): all dense GEMMs via v_wmma_f32_16x16x32_bf16 (wave32,
// fp32 accum). 128x128 block tile, 8 waves (2Mx4N), 64x32 per wave =
// 8 WMMA / K-step / wave. LDS is double-buffered and filled with
// global_load_async_to_lds_b128 (ASYNCcnt); s_wait_asynccnt 4 gates buffer
// reuse while the next tile's DMA stays in flight. Last K-tile is peeled so
// the steady-state loop is branch-free; unroll-by-2 constant-folds the
// double-buffer select into immediate LDS offsets.
// ---------------------------------------------------------------------------

typedef __attribute__((ext_vector_type(16))) __bf16 v16bf;
typedef __attribute__((ext_vector_type(8)))  float  v8f;

static constexpr int TM = 128, TN = 128, TK = 32, PAD = 8, LDT = TK + PAD;
static constexpr int ABUF = TM * LDT;   // elements per A buffer
static constexpr int BBUF = TN * LDT;   // elements per B buffer

__device__ __forceinline__ v16bf frag_ld(const __bf16* p) {
    // CDNA5 16-bit A/B fragment: 16B at K=kb..kb+7 (VGPR0..3),
    // 16B at K=kb+16..kb+23 (VGPR4..7).
    union { uint4 q[2]; v16bf v; } u;
    u.q[0] = *(const uint4*)(p);
    u.q[1] = *(const uint4*)(p + 16);
    return u.v;
}

// Async DMA one 16B chunk global -> LDS (per lane). ASYNCcnt-tracked.
__device__ __forceinline__ void async_ld16(unsigned ldsaddr, const __bf16* g) {
    asm volatile("global_load_async_to_lds_b128 %0, %1, off"
                 :: "v"(ldsaddr), "v"((unsigned long long)(uintptr_t)g)
                 : "memory");
}

// C = A[M,K] @ W[N,K]^T + bias, optional ReLU; writes f32 and/or bf16 outputs.
// Requires: N % 128 == 0, K % 32 == 0. M guarded (branchless clamp on loads,
// predicated stores).
__global__ __launch_bounds__(256) void gemm_bf16_wmma(
    const __bf16* __restrict__ A, const __bf16* __restrict__ W,
    const float* __restrict__ bias, float* __restrict__ Cf,
    __bf16* __restrict__ Cb, int M, int Nn, int K, int act)
{
    __shared__ __align__(16) __bf16 As[2 * ABUF];
    __shared__ __align__(16) __bf16 Bs[2 * BBUF];

    const int tid  = threadIdx.x;
    const int lane = tid & 31;
    const int wv   = tid >> 5;          // 8 waves
    const int wm   = wv >> 2;           // 2 waves along M (64 rows each)
    const int wn   = wv & 3;            // 4 waves along N (32 cols each)
    const int m0   = blockIdx.x * TM;
    const int n0   = blockIdx.y * TN;

    const int fr = lane & 15;           // row/col within 16-wide fragment
    const int kb = (lane >> 4) * 8;     // lanes 0-15: kb=0, lanes 16-31: kb=8

    // Per-thread staging slots: two 16B chunks for A, two for B per tile.
    const int r0 = tid >> 2;            // 0..63
    const int kk = (tid & 3) * 8;
    int grA0 = m0 + r0;       if (grA0 >= M) grA0 = M - 1;   // branchless clamp
    int grA1 = m0 + r0 + 64;  if (grA1 >= M) grA1 = M - 1;
    const __bf16* pA0 = A + (long)grA0 * K + kk;
    const __bf16* pA1 = A + (long)grA1 * K + kk;
    const __bf16* pB0 = W + (long)(n0 + r0) * K + kk;
    const __bf16* pB1 = W + (long)(n0 + r0 + 64) * K + kk;

    unsigned lA0[2], lA1[2], lB0[2], lB1[2];
    #pragma unroll
    for (int b = 0; b < 2; ++b) {
        lA0[b] = (unsigned)(uintptr_t)(As + b * ABUF + (r0)      * LDT + kk);
        lA1[b] = (unsigned)(uintptr_t)(As + b * ABUF + (r0 + 64) * LDT + kk);
        lB0[b] = (unsigned)(uintptr_t)(Bs + b * BBUF + (r0)      * LDT + kk);
        lB1[b] = (unsigned)(uintptr_t)(Bs + b * BBUF + (r0 + 64) * LDT + kk);
    }

    v8f acc[4][2] = {};

    // 8 WMMA on the tile sitting in buffer `cur`.
    auto compute_tile = [&](int cur) {
        const __bf16* Ab = As + cur * ABUF + (wm * 64 + fr) * LDT + kb;
        const __bf16* Bb = Bs + cur * BBUF + (wn * 32 + fr) * LDT + kb;
        v16bf b0 = frag_ld(Bb);
        v16bf b1 = frag_ld(Bb + 16 * LDT);
        #pragma unroll
        for (int i = 0; i < 4; ++i) {
            v16bf a = frag_ld(Ab + i * 16 * LDT);
            acc[i][0] = __builtin_amdgcn_wmma_f32_16x16x32_bf16(false, a, false, b0, (short)0, acc[i][0], false, false);
            acc[i][1] = __builtin_amdgcn_wmma_f32_16x16x32_bf16(false, a, false, b1, (short)0, acc[i][1], false, false);
        }
    };

    const int kTiles = K / TK;

    // Prologue: stage tile 0 into buffer 0.
    async_ld16(lA0[0], pA0); async_ld16(lA1[0], pA1);
    async_ld16(lB0[0], pB0); async_ld16(lB1[0], pB1);
    pA0 += TK; pA1 += TK; pB0 += TK; pB1 += TK;

    // Steady state (branch-free body): issue tile kt+1, wait on tile kt, compute.
    #pragma unroll 2
    for (int kt = 0; kt < kTiles - 1; ++kt) {
        const int cur = kt & 1;
        const int nxt = cur ^ 1;
        async_ld16(lA0[nxt], pA0); async_ld16(lA1[nxt], pA1);
        async_ld16(lB0[nxt], pB0); async_ld16(lB1[nxt], pB1);
        pA0 += TK; pA1 += TK; pB0 += TK; pB1 += TK;
        // Only the previous tile's 4 async ops must have completed.
        asm volatile("s_wait_asynccnt 0x4" ::: "memory");
        __syncthreads();
        compute_tile(cur);
        __syncthreads();   // all waves done reading buf[cur] before refill
    }

    // Peeled last tile.
    asm volatile("s_wait_asynccnt 0x0" ::: "memory");
    __syncthreads();
    compute_tile((kTiles - 1) & 1);

    // Epilogue. C/D layout: VGPR q -> row (lane<16 ? q : q+8), col = lane&15.
    #pragma unroll
    for (int i = 0; i < 4; ++i) {
        #pragma unroll
        for (int j = 0; j < 2; ++j) {
            int c  = n0 + wn * 32 + j * 16 + (lane & 15);
            int rb = m0 + wm * 64 + i * 16 + ((lane >> 4) * 8);
            float bv = bias ? bias[c] : 0.0f;
            #pragma unroll
            for (int q = 0; q < 8; ++q) {
                int r = rb + q;
                if (r < M) {
                    float v = acc[i][j][q] + bv;
                    if (act) v = v > 0.f ? v : 0.f;
                    if (Cf) Cf[(long)r * Nn + c] = v;
                    if (Cb) Cb[(long)r * Nn + c] = (__bf16)v;
                }
            }
        }
    }
}

// ---------------- elementwise / reduction helpers ----------------

__global__ void f2bf_kernel(const float* __restrict__ x, __bf16* __restrict__ y, long n) {
    long i = (long)blockIdx.x * blockDim.x + threadIdx.x;
    if (i < n) y[i] = (__bf16)x[i];
}

__global__ void zero_kernel(float* __restrict__ x, long n) {
    long i = (long)blockIdx.x * blockDim.x + threadIdx.x;
    if (i < n) x[i] = 0.f;
}

// LayerNorm(256) + ReLU -> bf16, one wave32 per row (8 elems/lane).
__global__ __launch_bounds__(256) void ln_relu_bf16_kernel(
    const float* __restrict__ X, const float* __restrict__ g,
    const float* __restrict__ b, __bf16* __restrict__ Y, int M)
{
    int lane = threadIdx.x & 31;
    int row  = blockIdx.x * 8 + (threadIdx.x >> 5);
    if (row >= M) return;
    const float* x = X + (long)row * 256;
    float v[8], s = 0.f;
    #pragma unroll
    for (int j = 0; j < 8; ++j) { v[j] = x[lane + 32 * j]; s += v[j]; }
    #pragma unroll
    for (int m = 16; m >= 1; m >>= 1) s += __shfl_xor(s, m, 32);
    float mu = s * (1.f / 256.f);
    float s2 = 0.f;
    #pragma unroll
    for (int j = 0; j < 8; ++j) { float d = v[j] - mu; s2 += d * d; }
    #pragma unroll
    for (int m = 16; m >= 1; m >>= 1) s2 += __shfl_xor(s2, m, 32);
    float inv = rsqrtf(s2 * (1.f / 256.f) + 1e-5f);
    #pragma unroll
    for (int j = 0; j < 8; ++j) {
        int c = lane + 32 * j;
        float y = (v[j] - mu) * inv * g[c] + b[c];
        y = y > 0.f ? y : 0.f;
        Y[(long)row * 256 + c] = (__bf16)y;
    }
}

// edge_in[e] = concat(h[src[e]], h[dst[e]], e_feat[e])  (bf16, 768 wide), 16B chunks
__global__ void gather_edge_in_kernel(const __bf16* __restrict__ h,
                                      const __bf16* __restrict__ e,
                                      const int* __restrict__ idx,
                                      __bf16* __restrict__ out, int E)
{
    long i = (long)blockIdx.x * blockDim.x + threadIdx.x;
    long total = (long)E * 96;            // 96 chunks of 8 bf16 per edge
    if (i >= total) return;
    int ed = (int)(i / 96), c = (int)(i % 96);
    const __bf16* p;
    if (c < 32)       p = h + (long)idx[ed] * 256 + c * 8;
    else if (c < 64)  p = h + (long)idx[E + ed] * 256 + (c - 32) * 8;
    else              p = e + (long)ed * 256 + (c - 64) * 8;
    *(uint4*)(out + i * 8) = *(const uint4*)p;
}

// pair[e] = concat(h[src[e]], h[dst[e]]) (bf16, 512 wide)
__global__ void gather_pair_kernel(const __bf16* __restrict__ h,
                                   const int* __restrict__ idx,
                                   __bf16* __restrict__ out, int E)
{
    long i = (long)blockIdx.x * blockDim.x + threadIdx.x;
    long total = (long)E * 64;
    if (i >= total) return;
    int ed = (int)(i >> 6), c = (int)(i & 63);
    const __bf16* p = (c < 32) ? h + (long)idx[ed] * 256 + c * 8
                               : h + (long)idx[E + ed] * 256 + (c - 32) * 8;
    *(uint4*)(out + i * 8) = *(const uint4*)p;
}

// out[m] = sigmoid(dot(X[m,:256], w) + b0) -- one wave32 per row
__global__ __launch_bounds__(256) void dot_sigmoid_kernel(
    const __bf16* __restrict__ X, const __bf16* __restrict__ w,
    const float* __restrict__ b, float* __restrict__ out, int M)
{
    int lane = threadIdx.x & 31;
    int row  = blockIdx.x * 8 + (threadIdx.x >> 5);
    if (row >= M) return;
    const __bf16* x = X + (long)row * 256;
    float s = 0.f;
    #pragma unroll
    for (int j = 0; j < 8; ++j) s += (float)x[lane + 32 * j] * (float)w[lane + 32 * j];
    #pragma unroll
    for (int m = 16; m >= 1; m >>= 1) s += __shfl_xor(s, m, 32);
    if (lane == 0) out[row] = 1.f / (1.f + expf(-(s + b[0])));
}

// agg[dst[e], c] += msg[e, c] * gate[e]   (hardware global_atomic_add_f32)
__global__ void scatter_kernel(const float* __restrict__ msg,
                               const float* __restrict__ gate,
                               const int* __restrict__ idx,
                               float* __restrict__ agg, int E)
{
    long i = (long)blockIdx.x * blockDim.x + threadIdx.x;
    if (i >= (long)E * 256) return;
    int e = (int)(i >> 8), c = (int)(i & 255);
    float v = msg[i] * gate[e];
    unsafeAtomicAdd(&agg[(long)idx[E + e] * 256 + c], v);
}

// GRU cell elementwise (gi/gh are [N,768], h is [N,256])
__global__ void gru_kernel(const float* __restrict__ gi, const float* __restrict__ gh,
                           float* __restrict__ h, __bf16* __restrict__ hb, int Nn)
{
    long i = (long)blockIdx.x * blockDim.x + threadIdx.x;
    if (i >= (long)Nn * 256) return;
    int n = (int)(i >> 8), c = (int)(i & 255);
    const float* gin = gi + (long)n * 768;
    const float* ghn = gh + (long)n * 768;
    float r  = 1.f / (1.f + expf(-(gin[c]       + ghn[c])));
    float z  = 1.f / (1.f + expf(-(gin[256 + c] + ghn[256 + c])));
    float nn = tanhf(gin[512 + c] + r * ghn[512 + c]);
    float hv = h[i];
    float out = (1.f - z) * nn + z * hv;
    h[i]  = out;
    hb[i] = (__bf16)out;
}

// ---------------------------------------------------------------------------

extern "C" void kernel_launch(void* const* d_in, const int* in_sizes, int n_in,
                              void* d_out, int out_size, void* d_ws, size_t ws_size,
                              hipStream_t stream) {
    (void)in_sizes; (void)n_in; (void)out_size; (void)ws_size;
    const int N_ = 10000, E_ = 160000;

    // ---- flattened input indices (setup_inputs dict order) ----
    const float* node_f = (const float*)d_in[0];
    const int*   eidx   = (const int*)d_in[2];     // [2,E]: src then dst
    const float* edge_f = (const float*)d_in[3];
    auto F = [&](int i) { return (const float*)d_in[i]; };
    const int L0 = 16, LS = 12;  // layer param base / stride
    const int L_M1B=1, L_LNG=2, L_LNB=3, L_M2B=5, L_GB=7, L_BIH=10, L_BHH=11;

    // ---- workspace bump allocator ----
    char* wsb = (char*)d_ws;
    size_t off = 0;
    auto alloc = [&](size_t bytes) -> void* {
        void* p = wsb + off;
        off += (bytes + 255) & ~(size_t)255;
        return p;
    };
    auto bf = [&](long n) { return (__bf16*)alloc((size_t)n * 2); };
    auto fl = [&](long n) { return (float*)alloc((size_t)n * 4); };

    // bf16 weights
    __bf16* w_ne1 = bf(256 * 256); __bf16* w_ne2 = bf(256 * 256);
    __bf16* w_ee1 = bf(256 * 128); __bf16* w_ee2 = bf(256 * 256);
    __bf16 *w_m1[3], *w_m2[3], *w_g[3], *w_ih[3], *w_hh[3];
    for (int l = 0; l < 3; ++l) {
        w_m1[l] = bf(256 * 768); w_m2[l] = bf(256 * 256); w_g[l] = bf(256);
        w_ih[l] = bf(768 * 256); w_hh[l] = bf(768 * 256);
    }
    __bf16* w_p1 = bf(256 * 512); __bf16* w_p2 = bf(256); __bf16* w_cls = bf(256 * 256);

    // activations
    __bf16* nf_bf   = bf((long)N_ * 256);
    __bf16* ef_bf   = bf((long)E_ * 128);
    float*  h_f     = fl((long)N_ * 256);
    __bf16* h_bf    = bf((long)N_ * 256);
    float*  tn_f    = fl((long)N_ * 256);
    __bf16* tn_bf   = bf((long)N_ * 256);
    __bf16* e_bf    = bf((long)E_ * 256);
    float*  te_f    = fl((long)E_ * 256);   // reused: enc t1 / layer t1 / msg
    __bf16* te_bf   = bf((long)E_ * 256);   // reused: LN outputs / p1 bf16
    __bf16* edge_in = bf((long)E_ * 768);   // reused as pair [E,512]
    float*  gatev   = fl(E_);
    float*  agg     = fl((long)N_ * 256);
    __bf16* agg_bf  = bf((long)N_ * 256);
    float*  gi      = fl((long)N_ * 768);
    float*  gh      = fl((long)N_ * 768);

    float* out_updated = (float*)d_out;                   // [N, 256]
    float* out_scores  = (float*)d_out + (long)N_ * 256;  // [E]

    // ---- launch helpers ----
    auto conv = [&](const float* x, __bf16* y, long n) {
        f2bf_kernel<<<dim3((unsigned)((n + 255) / 256)), dim3(256), 0, stream>>>(x, y, n);
    };
    auto gemm = [&](const __bf16* A, const __bf16* W, const float* bias,
                    float* Cf, __bf16* Cb, int M, int Nn, int K, int act) {
        dim3 g((M + TM - 1) / TM, Nn / TN);
        gemm_bf16_wmma<<<g, dim3(256), 0, stream>>>(A, W, bias, Cf, Cb, M, Nn, K, act);
    };
    auto ln = [&](const float* X, const float* g, const float* b, __bf16* Y, int M) {
        ln_relu_bf16_kernel<<<dim3((M + 7) / 8), dim3(256), 0, stream>>>(X, g, b, Y, M);
    };

    // ---- convert weights to bf16 ----
    conv(F(4),  w_ne1, 256 * 256);  conv(F(8),  w_ne2, 256 * 256);
    conv(F(10), w_ee1, 256 * 128);  conv(F(14), w_ee2, 256 * 256);
    for (int l = 0; l < 3; ++l) {
        int b0 = L0 + l * LS;
        conv(F(b0 + 0), w_m1[l], 256 * 768);
        conv(F(b0 + 4), w_m2[l], 256 * 256);
        conv(F(b0 + 6), w_g[l],  256);
        conv(F(b0 + 8), w_ih[l], 768 * 256);
        conv(F(b0 + 9), w_hh[l], 768 * 256);
    }
    conv(F(52), w_p1, 256 * 512); conv(F(54), w_p2, 256); conv(F(56), w_cls, 256 * 256);

    // ---- encoders ----
    conv(node_f, nf_bf, (long)N_ * 256);
    conv(edge_f, ef_bf, (long)E_ * 128);
    // node: Linear -> LN+ReLU -> Linear
    gemm(nf_bf, w_ne1, F(5), tn_f, nullptr, N_, 256, 256, 0);
    ln(tn_f, F(6), F(7), tn_bf, N_);
    gemm(tn_bf, w_ne2, F(9), h_f, h_bf, N_, 256, 256, 0);
    // edge: Linear(128->256) -> LN+ReLU -> Linear
    gemm(ef_bf, w_ee1, F(11), te_f, nullptr, E_, 256, 128, 0);
    ln(te_f, F(12), F(13), te_bf, E_);
    gemm(te_bf, w_ee2, F(15), nullptr, e_bf, E_, 256, 256, 0);

    // ---- message-passing layers ----
    for (int l = 0; l < 3; ++l) {
        int b0 = L0 + l * LS;
        gather_edge_in_kernel<<<dim3((unsigned)(((long)E_ * 96 + 255) / 256)),
                                dim3(256), 0, stream>>>(h_bf, e_bf, eidx, edge_in, E_);
        gemm(edge_in, w_m1[l], F(b0 + L_M1B), te_f, nullptr, E_, 256, 768, 0);
        ln(te_f, F(b0 + L_LNG), F(b0 + L_LNB), te_bf, E_);
        gemm(te_bf, w_m2[l], F(b0 + L_M2B), te_f, nullptr, E_, 256, 256, 0);  // msg
        dot_sigmoid_kernel<<<dim3((E_ + 7) / 8), dim3(256), 0, stream>>>(
            e_bf, w_g[l], F(b0 + L_GB), gatev, E_);
        zero_kernel<<<dim3((unsigned)(((long)N_ * 256 + 255) / 256)), dim3(256), 0, stream>>>(
            agg, (long)N_ * 256);
        scatter_kernel<<<dim3((unsigned)(((long)E_ * 256 + 255) / 256)), dim3(256), 0, stream>>>(
            te_f, gatev, eidx, agg, E_);
        conv(agg, agg_bf, (long)N_ * 256);
        gemm(agg_bf, w_ih[l], F(b0 + L_BIH), gi, nullptr, N_, 768, 256, 0);
        gemm(h_bf,   w_hh[l], F(b0 + L_BHH), gh, nullptr, N_, 768, 256, 0);
        gru_kernel<<<dim3((unsigned)(((long)N_ * 256 + 255) / 256)), dim3(256), 0, stream>>>(
            gi, gh, h_f, h_bf, N_);
    }

    // ---- edge score head ----
    gather_pair_kernel<<<dim3((unsigned)(((long)E_ * 64 + 255) / 256)), dim3(256), 0, stream>>>(
        h_bf, eidx, edge_in, E_);
    gemm(edge_in, w_p1, F(53), nullptr, te_bf, E_, 256, 512, 1);  // ReLU fused
    dot_sigmoid_kernel<<<dim3((E_ + 7) / 8), dim3(256), 0, stream>>>(
        te_bf, w_p2, F(55), out_scores, E_);

    // ---- node classifier head ----
    gemm(h_bf, w_cls, F(57), out_updated, nullptr, N_, 256, 256, 0);
}
